// AttLoc_85598698209863
// MI455X (gfx1250) — matrix operand order repaired and checked by reading, not compiled, not run on hardware
//
#include <hip/hip_runtime.h>
#include <math.h>

typedef __attribute__((ext_vector_type(16))) __bf16 v16bf;
typedef __attribute__((ext_vector_type(8)))  __bf16 v8bf;
typedef __attribute__((ext_vector_type(8)))  float  v8f;

#define B_  32
#define T_  2048
#define E_  1024
#define A_  512
#define C_  32
#define KW_ 31

__device__ __forceinline__ __bf16 tobf(float x) { return (__bf16)x; }

// ---------------------------------------------------------------------------
// Prep 1: dec_t[b][a] = dec_z[b] . W_dec[:,a] + b_enc[a]   (fp32, 32x512)
// ---------------------------------------------------------------------------
__global__ __launch_bounds__(256) void prep_dec(const float* __restrict__ dec_z,
                                                const float* __restrict__ W_dec,
                                                const float* __restrict__ b_enc,
                                                float* __restrict__ dect) {
    int idx = blockIdx.x * 256 + threadIdx.x;      // 16384 threads
    int b = idx >> 9, a = idx & 511;
    float s = b_enc[a];
    const float* z = dec_z + b * 1024;
    const float* w = W_dec + a;
    for (int k = 0; k < 1024; ++k) s += z[k] * w[(size_t)k * A_];
    dect[idx] = s;
}

// ---------------------------------------------------------------------------
// Prep 2: WencT[a][k] = bf16(W_enc[k][a])    (512 x 1024 bf16)
// ---------------------------------------------------------------------------
__global__ __launch_bounds__(256) void prep_wencT(const float* __restrict__ W_enc,
                                                  __bf16* __restrict__ WencT) {
    int idx = blockIdx.x * 256 + threadIdx.x;      // 524288 threads
    int k = idx >> 9, a = idx & 511;
    WencT[(size_t)a * E_ + k] = tobf(W_enc[idx]);
}

// ---------------------------------------------------------------------------
// Prep 3: WattT[a][c] = bf16(W_att[c][a])    (512 x 32 bf16)
// ---------------------------------------------------------------------------
__global__ __launch_bounds__(256) void prep_wattT(const float* __restrict__ W_att,
                                                  __bf16* __restrict__ WattT) {
    int idx = blockIdx.x * 256 + threadIdx.x;      // 16384 threads
    int c = idx >> 9, a = idx & 511;
    WattT[a * C_ + c] = tobf(W_att[idx]);
}

// ---------------------------------------------------------------------------
// Prep 4: location conv, SAME padding:  convA[b][t][c] bf16
// ---------------------------------------------------------------------------
__global__ __launch_bounds__(256) void prep_conv(const float* __restrict__ att_prev,
                                                 const float* __restrict__ conv_w,
                                                 __bf16* __restrict__ convA) {
    int idx = blockIdx.x * 256 + threadIdx.x;      // 32*2048*32 threads
    int c = idx & 31;
    int t = (idx >> 5) & 2047;
    int b = idx >> 16;
    const float* ap = att_prev + b * T_;
    const float* cw = conv_w + c * KW_;
    float s = 0.f;
    #pragma unroll
    for (int j = 0; j < KW_; ++j) {
        int tp = t + j - (KW_ / 2);
        if (tp >= 0 && tp < T_) s += ap[tp] * cw[j];
    }
    convA[idx] = tobf(s);
}

// ---------------------------------------------------------------------------
// Main fused energy kernel: per WG, 64 rows of (b,t), full A=512 via 4 chunks.
//   acc = WMMA(convA_tile, WattT)            (K=32, att term)
//   acc += sum_kc WMMA(enc_tile, WencT)      (K=1024)
//   e[row] = sum_a tanh(acc + dec_t[b][a]) * W_g[a]
// ---------------------------------------------------------------------------
__global__ __launch_bounds__(256)
void att_energy_kernel(const float*  __restrict__ enc,    // (B,T,E) fp32
                       const float*  __restrict__ dect,   // (B,A)  fp32 (incl b_enc)
                       const __bf16* __restrict__ WencT,  // (A,E)  bf16
                       const __bf16* __restrict__ WattT,  // (A,C)  bf16
                       const __bf16* __restrict__ convA,  // (B,T,C) bf16
                       const float*  __restrict__ Wg,     // (A)
                       float* __restrict__ eOut)          // (B,T)
{
    __shared__ __bf16 encS[64 * 1032];   // padded stride -> conflict-free b128 gathers
    __shared__ float  decS[A_];
    __shared__ float  wgS[A_];
    __shared__ float  eS[64];

    const int tid  = threadIdx.x;
    const int lane = tid & 31;
    const int wave = tid >> 5;               // 0..7 -> N-tile within chunk
    const int b    = blockIdx.y;
    const int t0   = blockIdx.x * 64;
    const int l15  = lane & 15;
    const int half = lane >> 4;

    for (int i = tid; i < A_; i += 256) {
        decS[i] = dect[b * A_ + i];
        wgS[i]  = Wg[i];
    }
    if (tid < 64) eS[tid] = 0.0f;

    // stage enc tile (64 x 1024) fp32 -> bf16 in LDS
    {
        const float* src = enc + ((size_t)b * T_ + t0) * E_;
        for (int i = tid; i < 64 * 256; i += 256) {
            int row = i >> 8;              // 256 float4 per row
            int c4  = i & 255;
            float4 v = *(const float4*)(src + (size_t)row * E_ + c4 * 4);
            __bf16* d = &encS[row * 1032 + c4 * 4];
            d[0] = tobf(v.x); d[1] = tobf(v.y); d[2] = tobf(v.z); d[3] = tobf(v.w);
        }
    }
    __syncthreads();

    for (int nc = 0; nc < 4; ++nc) {
        const int n = nc * 128 + wave * 16 + l15;   // output column 0..511

        // B fragment for the att GEMM: WattT row n, K = half*8 .. +7 and 16 + half*8 .. +7
        v16bf battw;
        {
            const v8bf* p = (const v8bf*)(WattT + (size_t)n * C_ + half * 8);
            v8bf lo = p[0], hi = p[2];
            #pragma unroll
            for (int i = 0; i < 8; ++i) { battw[i] = lo[i]; battw[8 + i] = hi[i]; }
        }

        v8f acc[4];
        // init accumulators with location-attention contribution (one K=32 WMMA each)
        #pragma unroll
        for (int mt = 0; mt < 4; ++mt) {
            const int row = t0 + mt * 16 + l15;
            v16bf afrag;
            {
                const v8bf* p = (const v8bf*)(convA + ((size_t)b * T_ + row) * C_ + half * 8);
                v8bf lo = p[0];   // k = half*8 .. +7
                v8bf hi = p[2];   // k = 16 + half*8 .. +7
                #pragma unroll
                for (int i = 0; i < 8; ++i) { afrag[i] = lo[i]; afrag[8 + i] = hi[i]; }
            }
            v8f z = {};
            acc[mt] = __builtin_amdgcn_wmma_f32_16x16x32_bf16(
                false, afrag, false, battw, (short)0, z, false, false);
        }

        // main K loop over encoder dim (1024 = 32 steps of K=32)
        for (int kc = 0; kc < 32; ++kc) {
            v16bf bfrag;
            {
                const v8bf* p = (const v8bf*)(WencT + (size_t)n * E_ + kc * 32 + half * 8);
                v8bf lo = p[0], hi = p[2];
                #pragma unroll
                for (int i = 0; i < 8; ++i) { bfrag[i] = lo[i]; bfrag[8 + i] = hi[i]; }
            }
            #pragma unroll
            for (int mt = 0; mt < 4; ++mt) {
                v16bf afrag;
                const v8bf* p = (const v8bf*)(&encS[(mt * 16 + l15) * 1032 + kc * 32 + half * 8]);
                v8bf lo = p[0];
                v8bf hi = p[2];
                #pragma unroll
                for (int i = 0; i < 8; ++i) { afrag[i] = lo[i]; afrag[8 + i] = hi[i]; }
                acc[mt] = __builtin_amdgcn_wmma_f32_16x16x32_bf16(
                    false, afrag, false, bfrag, (short)0, acc[mt], false, false);
            }
        }

        // epilogue: tanh, weight by W_g, reduce into per-row energy
        const float dv = decS[n];
        const float wg = wgS[n];
        #pragma unroll
        for (int mt = 0; mt < 4; ++mt) {
            #pragma unroll
            for (int r = 0; r < 8; ++r) {
                float val = tanhf(acc[mt][r] + dv) * wg;
                // C layout: VGPR r -> row mt*16 + r + 8*half, lane&15 -> col
                val += __shfl_xor(val, 1, 16);
                val += __shfl_xor(val, 2, 16);
                val += __shfl_xor(val, 4, 16);
                val += __shfl_xor(val, 8, 16);
                if (l15 == 0) {
                    int m = mt * 16 + r + 8 * half;
                    atomicAdd(&eS[m], val);
                }
            }
        }
    }
    __syncthreads();
    if (tid < 64) eOut[(size_t)b * T_ + t0 + tid] = eS[tid];
}

// ---------------------------------------------------------------------------
// Masked, scaled softmax over T per batch row. Writes w into d_out, zeros c.
// ---------------------------------------------------------------------------
__global__ __launch_bounds__(256)
void softmax_kernel(const float* __restrict__ e, const int* __restrict__ lenv,
                    const float* __restrict__ b_g, float* __restrict__ out) {
    int b = blockIdx.x, tid = threadIdx.x;
    __shared__ float red[256];
    int L = lenv[b];
    float bg = b_g[0];
    float v[8];
    float lmax = -1e30f;
    #pragma unroll
    for (int i = 0; i < 8; ++i) {
        int t = tid + i * 256;
        float x = (t < L) ? 2.0f * (e[b * T_ + t] + bg) : -1e30f;
        v[i] = x;
        lmax = fmaxf(lmax, x);
    }
    red[tid] = lmax; __syncthreads();
    for (int s = 128; s > 0; s >>= 1) {
        if (tid < s) red[tid] = fmaxf(red[tid], red[tid + s]);
        __syncthreads();
    }
    float m = red[0]; __syncthreads();
    float lsum = 0.f;
    #pragma unroll
    for (int i = 0; i < 8; ++i) {
        int t = tid + i * 256;
        float ex = (t < L) ? __expf(v[i] - m) : 0.0f;
        v[i] = ex; lsum += ex;
    }
    red[tid] = lsum; __syncthreads();
    for (int s = 128; s > 0; s >>= 1) {
        if (tid < s) red[tid] += red[tid + s];
        __syncthreads();
    }
    float inv = 1.0f / red[0];
    #pragma unroll
    for (int i = 0; i < 8; ++i) {
        int t = tid + i * 256;
        out[B_ * E_ + b * T_ + t] = v[i] * inv;   // w region after c (32*1024)
    }
    // zero the context region for this batch row (context kernel atomically adds)
    for (int i = tid; i < E_; i += 256) out[b * E_ + i] = 0.0f;
}

// ---------------------------------------------------------------------------
// Context: c[b][d] += sum_t w[b][t] * enc[b][t][d], T split across 8 segments.
// ---------------------------------------------------------------------------
__global__ __launch_bounds__(256)
void context_kernel(const float* __restrict__ enc, const float* __restrict__ w,
                    float* __restrict__ out_c) {
    __shared__ float sw[256];
    int tid = threadIdx.x;
    int dc = blockIdx.x;   // 0..3
    int ts = blockIdx.y;   // 0..7
    int b  = blockIdx.z;   // 0..31
    sw[tid] = w[b * T_ + ts * 256 + tid];
    __syncthreads();
    int d = dc * 256 + tid;
    const float* ep = enc + ((size_t)b * T_ + ts * 256) * E_ + d;
    float acc = 0.f;
    #pragma unroll 4
    for (int i = 0; i < 256; ++i) acc += sw[i] * ep[(size_t)i * E_];
    atomicAdd(&out_c[b * E_ + d], acc);
}

// ---------------------------------------------------------------------------
extern "C" void kernel_launch(void* const* d_in, const int* in_sizes, int n_in,
                              void* d_out, int out_size, void* d_ws, size_t ws_size,
                              hipStream_t stream) {
    const float* enc     = (const float*)d_in[0];
    const int*   enclen  = (const int*)d_in[1];
    const float* dec_z   = (const float*)d_in[2];
    const float* attprev = (const float*)d_in[3];
    const float* W_enc   = (const float*)d_in[4];
    const float* b_enc   = (const float*)d_in[5];
    const float* W_dec   = (const float*)d_in[6];
    const float* W_att   = (const float*)d_in[7];
    const float* conv_w  = (const float*)d_in[8];
    const float* W_g     = (const float*)d_in[9];
    const float* b_g     = (const float*)d_in[10];
    float* out = (float*)d_out;

    char* ws = (char*)d_ws;
    float*  eW    = (float*)(ws);                                     // 256 KB
    float*  dect  = (float*)(ws + 262144);                            // 64 KB
    __bf16* WencT = (__bf16*)(ws + 262144 + 65536);                   // 1 MB
    __bf16* WattT = (__bf16*)(ws + 262144 + 65536 + 1048576);         // 32 KB
    __bf16* convA = (__bf16*)(ws + 262144 + 65536 + 1048576 + 32768); // 4 MB

    prep_dec  <<<64,   256, 0, stream>>>(dec_z, W_dec, b_enc, dect);
    prep_wencT<<<2048, 256, 0, stream>>>(W_enc, WencT);
    prep_wattT<<<64,   256, 0, stream>>>(W_att, WattT);
    prep_conv <<<8192, 256, 0, stream>>>(attprev, conv_w, convA);

    dim3 g(T_ / 64, B_);   // (32 t-tiles, 32 batch)
    att_energy_kernel<<<g, 256, 0, stream>>>(enc, dect, WencT, WattT, convA, W_g, eW);

    softmax_kernel<<<B_, 256, 0, stream>>>(eW, enclen, b_g, out);

    dim3 gc(4, 8, B_);
    context_kernel<<<gc, 256, 0, stream>>>(enc, out + B_ * E_, out);
}
